// SAM_64183991271995
// MI455X (gfx1250) — matrix operand order repaired and checked by reading, not compiled, not run on hardware
//
#include <hip/hip_runtime.h>
#include <cstddef>

// ---------------- problem constants ----------------
#define BN      4
#define CIN     256
#define HH      56
#define WW      56
#define SPAT    (HH * WW)           // 3136
#define NROWS   (BN * SPAT)         // 12544
#define RELC    16
#define OUTC    256
#define SPC     8
#define WSC     32                  // OUTC / SPC
#define K2      49
#define YC      288                 // 16 + 16 + 256 fused GEMM columns
#define NT      (NROWS / 16)        // 784 row tiles
#define CT      (YC / 16)           // 18 col tiles

typedef __attribute__((ext_vector_type(8)))  __bf16 v8bf;
typedef __attribute__((ext_vector_type(16))) __bf16 v16bf;
typedef __attribute__((ext_vector_type(8)))  float  v8f;

__device__ __forceinline__ unsigned short f32_to_bf16_rne(float f) {
    unsigned u = __float_as_uint(f);
    unsigned r = u + 0x7FFFu + ((u >> 16) & 1u);   // round-to-nearest-even
    return (unsigned short)(r >> 16);
}
__device__ __forceinline__ __bf16 tobf(float f) {
    unsigned short h = f32_to_bf16_rne(f);
    __bf16 r;
    __builtin_memcpy(&r, &h, 2);
    return r;
}
__device__ __forceinline__ v16bf cat16(v8bf lo, v8bf hi) {
    return __builtin_shufflevector(lo, hi, 0, 1, 2, 3, 4, 5, 6, 7,
                                           8, 9, 10, 11, 12, 13, 14, 15);
}

// -------- pack x [B,C,H,W] f32 -> Xt [N=B*H*W, 256] bf16 (coalesced reads) --------
__global__ void k_pack_x(const float* __restrict__ x, unsigned short* __restrict__ Xt) {
    int tid = blockIdx.x * 256 + threadIdx.x;      // 256 * 12544 threads total
    int c = tid / NROWS;
    int n = tid % NROWS;
    int b = n / SPAT, sp = n % SPAT;
    float v = x[((size_t)(b * CIN + c)) * SPAT + sp];
    Xt[(size_t)n * CIN + c] = f32_to_bf16_rne(v);
}

// -------- pack [w1|w2|w3] -> Wb [288][256] bf16 + Bias[288] f32 --------
__global__ void k_pack_w(const float* __restrict__ w1, const float* __restrict__ b1,
                         const float* __restrict__ w2, const float* __restrict__ b2,
                         const float* __restrict__ w3, const float* __restrict__ b3,
                         unsigned short* __restrict__ Wb, float* __restrict__ Bias) {
    int o = blockIdx.x;        // 0..287
    int c = threadIdx.x;       // 0..255
    float v = (o < 16) ? w1[o * CIN + c]
            : (o < 32) ? w2[(o - 16) * CIN + c]
                       : w3[(o - 32) * CIN + c];
    Wb[(size_t)o * CIN + c] = f32_to_bf16_rne(v);
    if (c == 0)
        Bias[o] = (o < 16) ? b1[o] : (o < 32) ? b2[o - 16] : b3[o - 32];
}

// -------- WMMA bf16 GEMM: Y[N,288] = Xt[N,256] * Wb^T + Bias --------
__global__ void __launch_bounds__(256) k_gemm(const unsigned short* __restrict__ Xt_,
                                              const unsigned short* __restrict__ Wb_,
                                              const float* __restrict__ Bias,
                                              float* __restrict__ Y) {
    const __bf16* Xt = reinterpret_cast<const __bf16*>(Xt_);
    const __bf16* Wb = reinterpret_cast<const __bf16*>(Wb_);

    int wave = threadIdx.y;                        // 0..7
    int lane = threadIdx.x;                        // 0..31
    int tile = blockIdx.x * 8 + wave;              // 784*18, exact
    int nt = tile / CT;
    int ct = tile % CT;

    int mn    = lane & 15;
    int khalf = lane >> 4;

    const __bf16* Arow = Xt + (size_t)(nt * 16 + mn) * CIN;
    const __bf16* Bcol = Wb + (size_t)(ct * 16 + mn) * CIN;

    v8f acc = {0.f, 0.f, 0.f, 0.f, 0.f, 0.f, 0.f, 0.f};

#pragma unroll
    for (int kb = 0; kb < CIN / 32; ++kb) {
        const __bf16* ap = Arow + kb * 32 + khalf * 16;
        v8bf alo = *reinterpret_cast<const v8bf*>(ap);
        v8bf ahi = *reinterpret_cast<const v8bf*>(ap + 8);
        v16bf a = cat16(alo, ahi);
        const __bf16* bp = Bcol + kb * 32 + khalf * 16;
        v16bf bm = *reinterpret_cast<const v16bf*>(bp);
        acc = __builtin_amdgcn_wmma_f32_16x16x32_bf16(
                  false, a, false, bm, (short)0, acc, false, false);
    }

    int ncol  = ct * 16 + mn;
    float bia = Bias[ncol];
    int rbase = nt * 16 + khalf * 8;
#pragma unroll
    for (int j = 0; j < 8; ++j)
        Y[(size_t)(rbase + j) * YC + ncol] = acc[j] + bia;
}

// -------- fused relation-MLP (WMMA) + softmax + aggregation + position conv --------
// one 64-thread (2-wave) block per pixel
__global__ void __launch_bounds__(64) k_attn(
        const float* __restrict__ Y,
        const float* __restrict__ bn1_g, const float* __restrict__ bn1_b,
        const float* __restrict__ bn1_m, const float* __restrict__ bn1_v,
        const float* __restrict__ cw1,
        const float* __restrict__ bn2_g, const float* __restrict__ bn2_b,
        const float* __restrict__ bn2_m, const float* __restrict__ bn2_v,
        const float* __restrict__ cw2, const float* __restrict__ cw2b,
        const float* __restrict__ pos2,
        float* __restrict__ out) {
    __shared__ float l_x1[16];
    __shared__ float l_s1[16], l_t1[16], l_s2[16], l_t2[16];
    __shared__ float l_cb[32];
    __shared__ float l_w[K2][WSC];                 // logits, then softmax weights
    __shared__ int   l_nbr[K2];
    __shared__ float l_agg[OUTC];
    // WMMA staging (bf16, K zero-padded 16 -> 32)
    __shared__ __align__(32) __bf16 sD[64 * 32];   // relu(bn1(diff)) taps x 16 (+pad)
    __shared__ __align__(32) __bf16 sU[64 * 32];   // relu(bn2(h1))   taps x 16 (+pad)
    __shared__ __align__(32) __bf16 sW1[16 * 32];  // cw1^T  [n][k] k-pad
    __shared__ __align__(32) __bf16 sW2[32 * 32];  // cw2^T  [n][k] k-pad

    int t  = threadIdx.x;              // 0..63
    int p  = blockIdx.x;               // pixel
    int b  = p / SPAT, sp = p % SPAT;
    int hy = sp / WW,  wx = sp % WW;

    if (t < 16) {
        float s1 = bn1_g[t] * rsqrtf(bn1_v[t] + 1e-5f);
        l_s1[t] = s1; l_t1[t] = bn1_b[t] - bn1_m[t] * s1;
        float s2 = bn2_g[t] * rsqrtf(bn2_v[t] + 1e-5f);
        l_s2[t] = s2; l_t2[t] = bn2_b[t] - bn2_m[t] * s2;
        l_x1[t] = Y[(size_t)p * YC + t];
    }
    if (t < 32) l_cb[t] = cw2b[t];
    for (int i = t; i < 16 * 32; i += 64) {        // cw1^T, K-padded
        int n = i >> 5, k = i & 31;
        sW1[i] = (k < 16) ? tobf(cw1[n * 16 + k]) : tobf(0.f);
    }
    for (int i = t; i < 32 * 32; i += 64) {        // cw2^T, K-padded
        int n = i >> 5, k = i & 31;
        sW2[i] = (k < 16) ? tobf(cw2[n * 16 + k]) : tobf(0.f);
    }
    for (int i = t; i < 64 * 16; i += 64)          // zero sU K-pad columns
        sU[(i >> 4) * 32 + 16 + (i & 15)] = tobf(0.f);
    if (t < K2) {                                  // reflect-padded neighbor rows
        int dy = t / 7 - 3, dx = t % 7 - 3;
        int h2 = hy + dy; h2 = (h2 < 0) ? -h2 : (h2 > HH - 1 ? 2 * HH - 2 - h2 : h2);
        int w2 = wx + dx; w2 = (w2 < 0) ? -w2 : (w2 > WW - 1 ? 2 * WW - 2 - w2 : w2);
        l_nbr[t] = b * SPAT + h2 * WW + w2;
    }
    __syncthreads();

    // ---- stage taps: diff -> BN1 -> ReLU -> bf16 rows of sD (one tap per thread)
    if (t < K2) {
        const float* yk = Y + (size_t)l_nbr[t] * YC + 16;
#pragma unroll
        for (int c = 0; c < 16; ++c) {
            float d = l_x1[c] - yk[c];
            d = d * l_s1[c] + l_t1[c];
            sD[t * 32 + c] = tobf(d > 0.f ? d : 0.f);
        }
    } else {
#pragma unroll
        for (int c = 0; c < 16; ++c) sD[t * 32 + c] = tobf(0.f);
    }
#pragma unroll
    for (int c = 16; c < 32; ++c) sD[t * 32 + c] = tobf(0.f);  // K-pad
    __syncthreads();

    int lane = t & 31, wv = t >> 5;
    int mn = lane & 15, kh = lane >> 4;

    // ---- GEMM1 (WMMA): H1[64x16] = sD[64x32] * cw1^T ; epilogue BN2+ReLU -> sU
    {
        v16bf bf = *reinterpret_cast<const v16bf*>(sW1 + mn * 32 + kh * 16);
        float s2 = l_s2[mn], t2v = l_t2[mn];
#pragma unroll
        for (int rr = 0; rr < 2; ++rr) {
            int rt = wv * 2 + rr;
            const __bf16* ap = sD + (rt * 16 + mn) * 32 + kh * 16;
            v16bf a = cat16(*reinterpret_cast<const v8bf*>(ap),
                            *reinterpret_cast<const v8bf*>(ap + 8));
            v8f acc = {0.f, 0.f, 0.f, 0.f, 0.f, 0.f, 0.f, 0.f};
            acc = __builtin_amdgcn_wmma_f32_16x16x32_bf16(
                      false, a, false, bf, (short)0, acc, false, false);
            int rbase = rt * 16 + kh * 8;          // D row = j + 8*kh (+tile), col = mn
#pragma unroll
            for (int j = 0; j < 8; ++j) {
                float h = acc[j] * s2 + t2v;
                sU[(rbase + j) * 32 + mn] = tobf(h > 0.f ? h : 0.f);
            }
        }
    }
    __syncthreads();

    // ---- GEMM2 (WMMA): logits[64x32] = sU[64x32] * cw2^T + cb -> l_w
    {
        int g = wv * 16 + mn;                      // this lane's output column
        v16bf bf = *reinterpret_cast<const v16bf*>(sW2 + g * 32 + kh * 16);
        float cb = l_cb[g];
#pragma unroll
        for (int rt = 0; rt < 4; ++rt) {
            const __bf16* ap = sU + (rt * 16 + mn) * 32 + kh * 16;
            v16bf a = cat16(*reinterpret_cast<const v8bf*>(ap),
                            *reinterpret_cast<const v8bf*>(ap + 8));
            v8f acc = {0.f, 0.f, 0.f, 0.f, 0.f, 0.f, 0.f, 0.f};
            acc = __builtin_amdgcn_wmma_f32_16x16x32_bf16(
                      false, a, false, bf, (short)0, acc, false, false);
            int rbase = rt * 16 + kh * 8;
#pragma unroll
            for (int j = 0; j < 8; ++j) {
                int row = rbase + j;
                if (row < K2) l_w[row][g] = acc[j] + cb;
            }
        }
    }
    __syncthreads();

    // ---- softmax over the 49 taps: one thread per weight-channel g
    if (t < WSC) {
        float m = -3.4e38f;
        for (int k = 0; k < K2; ++k) m = fmaxf(m, l_w[k][t]);
        float s = 0.f;
        for (int k = 0; k < K2; ++k) s += __expf(l_w[k][t] - m);
        float inv = 1.f / s;
        for (int k = 0; k < K2; ++k) l_w[k][t] = __expf(l_w[k][t] - m) * inv;
    }
    __syncthreads();

    // ---- aggregation: thread t owns value channels 4t..4t+3 (same g per thread)
    {
        int c0 = t * 4;
        int g  = c0 >> 3;
        float a0 = 0.f, a1 = 0.f, a2 = 0.f, a3 = 0.f;
        for (int k = 0; k < K2; ++k) {
            if (k + 1 < K2)
                __builtin_prefetch(Y + (size_t)l_nbr[k + 1] * YC + 32 + c0, 0, 0);
            const float4 xv = *reinterpret_cast<const float4*>(
                Y + (size_t)l_nbr[k] * YC + 32 + c0);
            float wk = l_w[k][g];
            a0 += wk * xv.x; a1 += wk * xv.y; a2 += wk * xv.z; a3 += wk * xv.w;
        }
        l_agg[c0] = a0; l_agg[c0 + 1] = a1; l_agg[c0 + 2] = a2; l_agg[c0 + 3] = a3;
    }
    __syncthreads();

    // ---- position conv: xc[g] = [agg interleave (8) | weights (49)]
    {
        int oc0 = t * 4;
        for (int j = 0; j < 4; ++j) {
            int oc = oc0 + j;
            int g = oc >> 3, o = oc & 7;
            const float* pw = pos2 + (size_t)(g * SPC + o) * (SPC + K2);
            float s = 0.f;
#pragma unroll
            for (int i = 0; i < SPC; ++i) s += l_agg[i * WSC + g] * pw[i];
            for (int k = 0; k < K2; ++k)  s += l_w[k][g] * pw[SPC + k];
            out[((size_t)(b * OUTC + oc)) * SPAT + sp] = s;
        }
    }
}

// ---------------- host side ----------------
static inline size_t align_up(size_t v, size_t a) { return (v + a - 1) & ~(a - 1); }

extern "C" void kernel_launch(void* const* d_in, const int* in_sizes, int n_in,
                              void* d_out, int out_size, void* d_ws, size_t ws_size,
                              hipStream_t stream) {
    const float* x     = (const float*)d_in[0];
    const float* w1    = (const float*)d_in[1];
    const float* b1    = (const float*)d_in[2];
    const float* w2    = (const float*)d_in[3];
    const float* b2    = (const float*)d_in[4];
    const float* w3    = (const float*)d_in[5];
    const float* b3    = (const float*)d_in[6];
    const float* bn1_g = (const float*)d_in[7];
    const float* bn1_b = (const float*)d_in[8];
    const float* bn1_m = (const float*)d_in[9];
    const float* bn1_v = (const float*)d_in[10];
    const float* cw1   = (const float*)d_in[11];
    const float* bn2_g = (const float*)d_in[12];
    const float* bn2_b = (const float*)d_in[13];
    const float* bn2_m = (const float*)d_in[14];
    const float* bn2_v = (const float*)d_in[15];
    const float* cw2   = (const float*)d_in[16];
    const float* cw2b  = (const float*)d_in[17];
    const float* pos2  = (const float*)d_in[18];
    float* out = (float*)d_out;

    // workspace carve-up (~21 MB total)
    char* ws = (char*)d_ws;
    size_t off = 0;
    unsigned short* Xt = (unsigned short*)(ws + off); off = align_up(off + (size_t)NROWS * CIN * 2, 256);
    unsigned short* Wb = (unsigned short*)(ws + off); off = align_up(off + (size_t)YC * CIN * 2, 256);
    float*          Bi = (float*)(ws + off);          off = align_up(off + (size_t)YC * 4, 256);
    float*          Yv = (float*)(ws + off);          off = align_up(off + (size_t)NROWS * YC * 4, 256);
    (void)ws_size; (void)in_sizes; (void)n_in; (void)out_size;

    k_pack_x<<<NROWS, 256, 0, stream>>>(x, Xt);
    k_pack_w<<<YC, 256, 0, stream>>>(w1, b1, w2, b2, w3, b3, Wb, Bi);
    k_gemm<<<(NT * CT) / 8, dim3(32, 8), 0, stream>>>(Xt, Wb, Bi, Yv);
    k_attn<<<NROWS, 64, 0, stream>>>(Yv,
                                     bn1_g, bn1_b, bn1_m, bn1_v, cw1,
                                     bn2_g, bn2_b, bn2_m, bn2_v, cw2, cw2b,
                                     pos2, out);
}
